// PairSimilarity_29205777613559
// MI455X (gfx1250) — matrix operand order repaired and checked by reading, not compiled, not run on hardware
//
#include <hip/hip_runtime.h>
#include <hip/hip_bf16.h>

// ---------------------------------------------------------------------------
// n1=n2=4096 pairs, natoms1=natoms2=1024 (fixed by setup_inputs).
// Output H[3072][3072] f32.  l=0.5 -> K2 = (4 - 16 u^2) * exp(-2 u^2).
// H = S1 * K2 * S2^T:  stage 1 gathers A1 = S1*K2 (sparse side, ~4 pairs/atom,
// 16.8M v_exp_f32); stage 2 is a dense bf16 WMMA GEMM, L2-resident (192MB L2),
// with async global->LDS double buffering for the shared B operand.
// ---------------------------------------------------------------------------
constexpr int N1  = 4096;
constexpr int N2  = 4096;
constexpr int NA1 = 1024;
constexpr int NA2 = 1024;
constexpr int RROWS = NA1 * 3;    // 3072
constexpr int CCOLS = NA2 * 3;    // 3072
constexpr int NT_N  = CCOLS / 16; // 192 column tiles
constexpr int NT_R  = RROWS / 16; // 192 row tiles
constexpr int KT    = N2 / 32;    // 128 k tiles

typedef __attribute__((ext_vector_type(16))) __bf16 v16bf;
typedef __attribute__((ext_vector_type(8)))  float  v8f;
typedef __attribute__((ext_vector_type(4)))  unsigned int v4u;
typedef __attribute__((ext_vector_type(4)))  int v4i;

// --- CDNA5 async global->LDS path (guarded; fallback = reg + ds_store) ------
#if __has_builtin(__builtin_amdgcn_global_load_async_to_lds_b128) && \
    __has_builtin(__builtin_amdgcn_s_wait_asynccnt)
#define USE_ASYNC 1
#else
#define USE_ASYNC 0
#endif

typedef __attribute__((address_space(1))) v4i gbl_v4i;
typedef __attribute__((address_space(3))) v4i lds_v4i;

__device__ __forceinline__ void stage16(const unsigned short* g, unsigned short* l) {
#if USE_ASYNC
  __builtin_amdgcn_global_load_async_to_lds_b128(
      (gbl_v4i*)(size_t)g, (lds_v4i*)(size_t)l, 0, 0);
#else
  *(v4u*)l = *(const v4u*)g;
#endif
}

__device__ __forceinline__ void stage_fence() {
#if USE_ASYNC
  __builtin_amdgcn_s_wait_asynccnt(0);
#endif
  __syncthreads();
}

__device__ __forceinline__ unsigned short f2bf(float f) {
  union { float f; unsigned int u; } v; v.f = f;
  unsigned int r = v.u + 0x7FFFu + ((v.u >> 16) & 1u);   // RNE
  return (unsigned short)(r >> 16);
}

// --- CSR over side-1 atoms (deterministic) ----------------------------------
__global__ void k_zero32(unsigned int* p, int n) {
  int i = blockIdx.x * blockDim.x + threadIdx.x;
  if (i < n) p[i] = 0u;
}

__global__ void k_zero128(v4u* p, long n16) {
  long i = (long)blockIdx.x * blockDim.x + threadIdx.x;
  if (i < n16) p[i] = (v4u){0u, 0u, 0u, 0u};
}

__global__ void k_count(const int* __restrict__ i1, int* __restrict__ counts, int n1) {
  int p = blockIdx.x * blockDim.x + threadIdx.x;
  if (p < n1) atomicAdd(counts + i1[p], 1);
}

__global__ void k_scan(const int* __restrict__ counts, int* __restrict__ offsets) {
  __shared__ int s[NA1];
  int t = threadIdx.x;
  s[t] = counts[t];
  __syncthreads();
  for (int off = 1; off < NA1; off <<= 1) {
    int v = (t >= off) ? s[t - off] : 0;
    __syncthreads();
    if (t >= off) s[t] += v;
    __syncthreads();
  }
  offsets[t + 1] = s[t];
  if (t == 0) offsets[0] = 0;
}

// stable fill: rank(p) = #{p' < p : i1[p'] == i1[p]} -> replay-deterministic sums
__global__ void k_rank_fill(const int* __restrict__ i1, const int* __restrict__ offsets,
                            int* __restrict__ list, int n1) {
  __shared__ int s[N1];
  for (int k = threadIdx.x; k < n1; k += blockDim.x) s[k] = i1[k];
  __syncthreads();
  int p = blockIdx.x * blockDim.x + threadIdx.x;
  if (p < n1) {
    int a = s[p], rank = 0;
    for (int pp = 0; pp < p; ++pp) rank += (s[pp] == a) ? 1 : 0;
    list[offsets[a] + rank] = p;
  }
}

// --- D2^T scattered directly into WMMA B-operand tile layout ----------------
// B tile (kt,nt): 1KB = 32 lanes x 32B. lane = 16*((k>>3)&1) + (n&15), e = (k&7) + 8*((k>>4)&1).
__global__ void k_scatter_d2t(const float* __restrict__ dr2, const int* __restrict__ i2,
                              unsigned short* __restrict__ D2T, int n2) {
  int q = blockIdx.x * blockDim.x + threadIdx.x;
  if (q >= n2) return;
  int kt = q >> 5;
  int klane = (q >> 3) & 1;
  int e = (q & 7) + (((q >> 4) & 1) << 3);
  int atom = i2[q];
  #pragma unroll
  for (int b = 0; b < 3; ++b) {
    int c = atom * 3 + b;
    int nt = c >> 4;
    int lane = klane * 16 + (c & 15);
    unsigned off = ((unsigned)(kt * NT_N + nt) << 9) + (unsigned)(lane << 4) + (unsigned)e;
    D2T[off] = f2bf(dr2[q * 3 + b]);
  }
}

// --- Stage 1: A1 = S1 * K2, written straight into A-operand tile layout -----
// A tile (rt,kt): 1KB. 16-bit A layout: lane = (r&15) + 16*((k>>3)&1),
// element e = (k&7) + 8*(k>>4)   (k = K within the 32-wide tile).
__global__ void __launch_bounds__(256)
k_build_a1(const float* __restrict__ d1, const float* __restrict__ dr1,
           const float* __restrict__ d2, const int* __restrict__ offsets,
           const int* __restrict__ list, unsigned short* __restrict__ A1) {
  int wave = (blockIdx.x * blockDim.x + threadIdx.x) >> 5;
  int lane = threadIdx.x & 31;
  const int nchunk = N2 >> 5;            // 128 (== KT)
  int m  = wave / nchunk;
  int kt = wave - m * nchunk;
  if (m >= NA1) return;
  int q = (kt << 5) + lane;
  float d2q = d2[q];
  float a0 = 0.f, a1 = 0.f, a2 = 0.f;
  int beg = offsets[m], end = offsets[m + 1];
  for (int t = beg; t < end; ++t) {
    int p = list[t];
    float u  = d1[p] - d2q;
    float u2 = u * u;
    float w  = (4.0f - 16.0f * u2) * __expf(-2.0f * u2);
    a0 = fmaf(dr1[p * 3 + 0], w, a0);
    a1 = fmaf(dr1[p * 3 + 1], w, a1);
    a2 = fmaf(dr1[p * 3 + 2], w, a2);
  }
  int kl = lane;                                   // K within tile
  int laneHi = ((kl >> 3) & 1) << 4;
  int e = (kl & 7) + ((kl >> 4) << 3);
  float acc[3] = {a0, a1, a2};
  #pragma unroll
  for (int a = 0; a < 3; ++a) {
    int r  = m * 3 + a;
    int rt = r >> 4;
    int ln = (r & 15) + laneHi;
    unsigned off = ((unsigned)(rt * KT + kt) << 9) + (unsigned)(ln << 4) + (unsigned)e;
    A1[off] = f2bf(acc[a]);
  }
}

// --- Stage 2: H = A1 x D2^T via bf16 WMMA -----------------------------------
// Block = 8 waves = 256 rows x 64 cols. Wave = 32x64 (2 A tiles x 4 B tiles,
// 8 WMMAs/k-step). B (4KB/k-step, shared by all 8 waves) is double-buffered in
// LDS via async global->LDS copies issued one k-step ahead; A is a contiguous
// per-wave v16bf stream prefetched one k-step ahead in registers. Last k-step
// is peeled so the steady-state loop has no conditional phis.
#define COMPUTE_STEP(CUR)                                                      \
  {                                                                            \
    const unsigned short* bbase = &sB[(CUR)][0] + (lane << 4);                 \
    v16bf b0 = *(const v16bf*)(bbase);                                         \
    v16bf b1 = *(const v16bf*)(bbase + 512);                                   \
    v16bf b2 = *(const v16bf*)(bbase + 1024);                                  \
    v16bf b3 = *(const v16bf*)(bbase + 1536);                                  \
    acc0 = __builtin_amdgcn_wmma_f32_16x16x32_bf16(false, a0, false, b0, (short)0, acc0, false, false); \
    acc4 = __builtin_amdgcn_wmma_f32_16x16x32_bf16(false, a1, false, b0, (short)0, acc4, false, false); \
    acc1 = __builtin_amdgcn_wmma_f32_16x16x32_bf16(false, a0, false, b1, (short)0, acc1, false, false); \
    acc5 = __builtin_amdgcn_wmma_f32_16x16x32_bf16(false, a1, false, b1, (short)0, acc5, false, false); \
    acc2 = __builtin_amdgcn_wmma_f32_16x16x32_bf16(false, a0, false, b2, (short)0, acc2, false, false); \
    acc6 = __builtin_amdgcn_wmma_f32_16x16x32_bf16(false, a1, false, b2, (short)0, acc6, false, false); \
    acc3 = __builtin_amdgcn_wmma_f32_16x16x32_bf16(false, a0, false, b3, (short)0, acc3, false, false); \
    acc7 = __builtin_amdgcn_wmma_f32_16x16x32_bf16(false, a1, false, b3, (short)0, acc7, false, false); \
  }

__global__ void __launch_bounds__(256)
k_gemm2(const unsigned short* __restrict__ A1, const unsigned short* __restrict__ D2T,
        float* __restrict__ out) {
  __shared__ unsigned short sB[2][2048];           // 2 x 4KB

  const int tid  = threadIdx.x;
  const int w    = tid >> 5;                        // wave in block, 0..7
  const int lane = tid & 31;
  const int NSTRIP = NT_N / 4;                      // 48
  const int rg   = blockIdx.x / NSTRIP;             // 0..11 (256-row group)
  const int nt0  = (blockIdx.x - rg * NSTRIP) * 4;  // 4 column tiles

  const int tA0 = rg * 16 + w * 2;                  // this wave's two A row tiles
  const unsigned short* aP0 = A1 + ((unsigned)(tA0 * KT) << 9) + (lane << 4);
  const unsigned short* aP1 = aP0 + ((unsigned)KT << 9);
  const unsigned short* gB = D2T + ((unsigned)nt0 << 9) + (w << 8) + (lane << 3);
  const unsigned gBstep = (unsigned)NT_N << 9;      // ushorts per k-step
  unsigned short* lB0 = &sB[0][0] + (w << 8) + (lane << 3);
  unsigned short* lB1 = &sB[1][0] + (w << 8) + (lane << 3);

  v8f acc0 = {}, acc1 = {}, acc2 = {}, acc3 = {};
  v8f acc4 = {}, acc5 = {}, acc6 = {}, acc7 = {};

  // prologue: stage k=0, prefetch A(0)
  stage16(gB, lB0);
  v16bf a0 = *(const v16bf*)aP0;
  v16bf a1 = *(const v16bf*)aP1;
  stage_fence();

  for (int kk = 0; kk < KT - 1; ++kk) {
    const int cur = kk & 1;
    // issue k+1 transfer + A prefetch under this step's WMMAs (unconditional)
    stage16(gB + (unsigned)(kk + 1) * gBstep, cur ? lB0 : lB1);
    v16bf a0n = *(const v16bf*)(aP0 + ((kk + 1) << 9));
    v16bf a1n = *(const v16bf*)(aP1 + ((kk + 1) << 9));

    COMPUTE_STEP(cur);

    a0 = a0n; a1 = a1n;
    stage_fence();   // k+1 transfer landed; all reads of sB[cur] done
  }
  // peeled final step (KT-1 is odd -> buffer 1)
  COMPUTE_STEP((KT - 1) & 1);

  // D layout: VGPR v: lanes 0-15 -> M=v, lanes 16-31 -> M=8+v; N = lane%16.
  const int hi = (lane >> 4) << 3;
  const int nb = lane & 15;
  v8f accA[4] = {acc0, acc1, acc2, acc3};
  v8f accB[4] = {acc4, acc5, acc6, acc7};
  #pragma unroll
  for (int j = 0; j < 4; ++j) {
    unsigned n = (unsigned)(nt0 + j) * 16 + nb;
    unsigned m0 = (unsigned)tA0 * 16 + hi;
    unsigned m1 = m0 + 16;
    #pragma unroll
    for (int v = 0; v < 8; ++v) {
      out[(m0 + v) * (unsigned)CCOLS + n] = accA[j][v];
      out[(m1 + v) * (unsigned)CCOLS + n] = accB[j][v];
    }
  }
}

// ---------------------------------------------------------------------------
extern "C" void kernel_launch(void* const* d_in, const int* in_sizes, int n_in,
                              void* d_out, int out_size, void* d_ws, size_t ws_size,
                              hipStream_t stream) {
  const float* d1  = (const float*)d_in[0];
  const float* dr1 = (const float*)d_in[1];
  const float* d2  = (const float*)d_in[2];
  const float* dr2 = (const float*)d_in[3];
  const int*   i1  = (const int*)d_in[4];
  const int*   i2  = (const int*)d_in[5];
  const int n1 = in_sizes[0];
  const int n2 = in_sizes[2];

  char* ws = (char*)d_ws;
  unsigned short* A1   = (unsigned short*)(ws);              // 25165824 B (A-op tiles)
  unsigned short* D2T  = (unsigned short*)(ws + 25165824);   // 25165824 B (B-op tiles)
  int* counts          = (int*)(ws + 50331648);
  int* offsets         = (int*)(ws + 50335744);
  int* list            = (int*)(ws + 50340096);
  float* out           = (float*)d_out;

  // 1) deterministic CSR over i1
  k_zero32<<<(NA1 + 255) / 256, 256, 0, stream>>>((unsigned int*)counts, NA1);
  k_count<<<(n1 + 255) / 256, 256, 0, stream>>>(i1, counts, n1);
  k_scan<<<1, NA1, 0, stream>>>(counts, offsets);
  k_rank_fill<<<(n1 + 1023) / 1024, 1024, 0, stream>>>(i1, offsets, list, n1);

  // 2) D2^T in B-operand tiles (zero, then scatter 12K nonzeros)
  {
    long n16 = (long)RROWS * N2 * 2 / 16;
    k_zero128<<<(unsigned)((n16 + 255) / 256), 256, 0, stream>>>((v4u*)D2T, n16);
  }
  k_scatter_d2t<<<(n2 + 255) / 256, 256, 0, stream>>>(dr2, i2, D2T, n2);

  // 3) A1 = S1 * K2 (16.8M exps), written in A-operand tiles
  {
    int waves = NA1 * (N2 / 32);
    k_build_a1<<<waves / 8, 256, 0, stream>>>(d1, dr1, d2, offsets, list, A1);
  }

  // 4) H = A1 x D2^T: async-LDS double-buffered bf16 WMMA GEMM
  {
    int blocks = (NT_R / 16) * (NT_N / 4);                   // 12 * 48 = 576
    k_gemm2<<<blocks, 256, 0, stream>>>(A1, D2T, out);
  }
}
